// PointerNet_35390530519907
// MI455X (gfx1250) — compile-verified, matrix operand order checked by reading, not run on hardware
//
#include <hip/hip_runtime.h>
#include <math.h>

// ---------------- problem constants ----------------
#define B        512
#define S        1024
#define H        128
#define G4       (4*H)      // 512 gate width
#define STEPS    5
#define GLIMPSES 5
#define NEGINF   (-1e9f)

typedef unsigned short bf16_t;  // raw bf16 bits in memory
typedef __attribute__((ext_vector_type(16))) __bf16 v16bf;
typedef __attribute__((ext_vector_type(8)))  float  v8f;
typedef int v4i_ __attribute__((vector_size(16)));

union FragB { v16bf v; uint4 u[2]; };

__device__ __forceinline__ float bf2f(bf16_t b) {
    unsigned u = ((unsigned)b) << 16;
    return __uint_as_float(u);
}
__device__ __forceinline__ bf16_t f2bf(float f) {
    unsigned u = __float_as_uint(f);
    unsigned r = u + 0x7FFFu + ((u >> 16) & 1u);   // round-to-nearest-even
    return (bf16_t)(r >> 16);
}
__device__ __forceinline__ float sigm(float x) { return 1.f / (1.f + __expf(-x)); }

// ---- CDNA5 async global->LDS copy (ASYNCcnt-tracked), guarded fallback ----
#if defined(__gfx1250__) && __has_builtin(__builtin_amdgcn_global_load_async_to_lds_b128)
#define HAS_ASYNC_LDS 1
#else
#define HAS_ASYNC_LDS 0
#endif

#if HAS_ASYNC_LDS
__device__ __forceinline__ void async_copy16(bf16_t* lds_dst, const bf16_t* gsrc) {
    v4i_* gg = (v4i_*)(void*)const_cast<bf16_t*>(gsrc);
    v4i_* ll = (v4i_*)(void*)lds_dst;
    __builtin_amdgcn_global_load_async_to_lds_b128(
        (__attribute__((address_space(1))) v4i_*)gg,
        (__attribute__((address_space(3))) v4i_*)ll,
        0, 0);
}
__device__ __forceinline__ void wait_async() {
#if __has_builtin(__builtin_amdgcn_s_wait_asynccnt)
    __builtin_amdgcn_s_wait_asynccnt(0);
#else
    asm volatile("s_wait_asynccnt 0x0" ::: "memory");
#endif
}
#endif

// Load a 16x32 bf16 WMMA A/B fragment from a row-major tile (row stride `ld` elems).
// Layout per CDNA5 ISA 7.12.2 (16-bit A 16x32): lanes 0-15: M=lane, K={kOff+0..7, kOff+16..23};
// lanes 16-31: M=lane-16, K={kOff+8..15, kOff+24..31}. B (32x16, W^T) mirrors with N in place
// of M when W is stored row-major [N,K].
__device__ __forceinline__ v16bf load_frag(const bf16_t* tile, int ld, int kOff, int lane) {
    int m  = lane & 15;
    int kb = kOff + ((lane & 16) ? 8 : 0);
    const bf16_t* p = tile + m * ld + kb;
    FragB f;
    f.u[0] = *(const uint4*)(p);
    f.u[1] = *(const uint4*)(p + 16);
    return f.v;
}

#define WMMA_BF16(A, Bf, C) \
    __builtin_amdgcn_wmma_f32_16x16x32_bf16(false, (A), false, (Bf), (short)0, (C), false, false)

// =====================================================================
// K1: 1x1-conv embedding  static[B,2,S] -> x_emb bf16 [B,S,H]
// (dynamic embedding of the reference is dead code w.r.t. outputs; skipped)
// =====================================================================
__global__ void embed_kernel(const float* __restrict__ st,
                             const float* __restrict__ W,   // [H,2]
                             const float* __restrict__ bias,// [H]
                             bf16_t* __restrict__ x_emb) {
    size_t idx = (size_t)blockIdx.x * blockDim.x + threadIdx.x;
    if (idx >= (size_t)B * S * H) return;
    int    h  = (int)(idx & (H - 1));
    size_t bs = idx / H;
    int    s  = (int)(bs & (S - 1));
    int    b  = (int)(bs / S);
    float v = W[h * 2 + 0] * st[((size_t)b * 2 + 0) * S + s]
            + W[h * 2 + 1] * st[((size_t)b * 2 + 1) * S + s]
            + bias[h];
    x_emb[idx] = f2bf(v);
}

// =====================================================================
// K2: LSTM encoder scan. 32 blocks x 16 batch rows; 256 thr = 8 waves.
// Weights (bf16, 256KB) resident in LDS for all 1024 steps; x tiles are
// double-buffered and staged with GLOBAL_LOAD_ASYNC_TO_LDS (ASYNCcnt).
// LDS layout (dynamic, 307200 B):
//   [0      ) sWih bf16 [4H][H]   131072
//   [131072 ) sWhh bf16 [4H][H]   131072
//   [262144 ) sX0  bf16 [16][H]     4096
//   [266240 ) sX1  bf16 [16][H]     4096
//   [270336 ) sH   bf16 [16][H]     4096
//   [274432 ) sG   f32  [16][4H]   32768
// =====================================================================
#define SMEM_ENC 307200

__global__ __launch_bounds__(256) void lstm_encoder_kernel(
    const bf16_t* __restrict__ x_emb,   // [B,S,H]
    const float*  __restrict__ Wih,     // [4H,H]
    const float*  __restrict__ Whh,     // [4H,H]
    const float*  __restrict__ bih,
    const float*  __restrict__ bhh,
    bf16_t* __restrict__ enc_outs,      // [B,S,H]
    float*  __restrict__ h_final,       // [B,H]
    float*  __restrict__ c_final) {     // [B,H]
    extern __shared__ char smem[];
    bf16_t* sWih = (bf16_t*)(smem);
    bf16_t* sWhh = (bf16_t*)(smem + 131072);
    bf16_t* sX0  = (bf16_t*)(smem + 262144);
    bf16_t* sX1  = (bf16_t*)(smem + 266240);
    bf16_t* sH   = (bf16_t*)(smem + 270336);
    float*  sG   = (float*) (smem + 274432);

    const int tid  = threadIdx.x;
    const int wave = tid >> 5, lane = tid & 31;
    const int b0   = blockIdx.x * 16;

    // elementwise ownership: row r in tile, hidden [j0, j0+8)
    const int r  = tid >> 4;
    const int j0 = (tid & 15) * 8;
    const bf16_t* xthread = x_emb + ((size_t)(b0 + r) * S) * H + j0;

    // prologue: stage x tile for t=0 into sX0
#if HAS_ASYNC_LDS
    async_copy16(sX0 + r * H + j0, xthread);
#else
    *(uint4*)(sX0 + r * H + j0) = *(const uint4*)xthread;
#endif

    for (int i = tid; i < G4 * H; i += 256) {
        sWih[i] = f2bf(Wih[i]);
        sWhh[i] = f2bf(Whh[i]);
    }
    for (int i = tid; i < 16 * H; i += 256) sH[i] = 0;

    float c[8], hv[8], bI[8], bF[8], bG[8], bO[8];
#pragma unroll
    for (int i = 0; i < 8; ++i) {
        int j = j0 + i;
        c[i] = 0.f; hv[i] = 0.f;
        bI[i] = bih[0 * H + j] + bhh[0 * H + j];
        bF[i] = bih[1 * H + j] + bhh[1 * H + j];
        bG[i] = bih[2 * H + j] + bhh[2 * H + j];
        bO[i] = bih[3 * H + j] + bhh[3 * H + j];
    }

    for (int t = 0; t < S; ++t) {
        bf16_t* sXc = (t & 1) ? sX1 : sX0;   // tile being consumed
        bf16_t* sXn = (t & 1) ? sX0 : sX1;   // tile being filled for t+1
#if HAS_ASYNC_LDS
        wait_async();                         // x tile for t has landed
#endif
        __syncthreads();                      // publish sXc / sH across waves

        // kick off the t+1 tile before the GEMM so its latency hides
        if (t + 1 < S) {
            const bf16_t* xnext = xthread + (size_t)(t + 1) * H;
#if HAS_ASYNC_LDS
            async_copy16(sXn + r * H + j0, xnext);
#else
            __builtin_prefetch(xnext + H, 0, 3);   // t+2, near scope
            *(uint4*)(sXn + r * H + j0) = *(const uint4*)xnext;
#endif
        }

        // ---- gates = X*Wih^T + Hprev*Whh^T : bf16 WMMA, f32 accum ----
        v16bf ax[4], ah[4];
#pragma unroll
        for (int ks = 0; ks < 4; ++ks) {
            ax[ks] = load_frag(sXc, H, ks * 32, lane);
            ah[ks] = load_frag(sH,  H, ks * 32, lane);
        }
#pragma unroll
        for (int nt = 0; nt < 4; ++nt) {
            const int n0 = wave * 64 + nt * 16;
            // two independent accumulation chains -> interleaved XDL issue
            v8f accX = {};
            v8f accH = {};
#pragma unroll
            for (int ks = 0; ks < 4; ++ks) {
                v16bf bx = load_frag(sWih + n0 * H, H, ks * 32, lane);
                v16bf bh = load_frag(sWhh + n0 * H, H, ks * 32, lane);
                accX = WMMA_BF16(ax[ks], bx, accX);
                accH = WMMA_BF16(ah[ks], bh, accH);
            }
            // C layout: VGPR v -> row (v + 8*(lane>=16)), col n0 + (lane&15)
            const int col   = n0 + (lane & 15);
            const int rbase = (lane >> 4) * 8;
#pragma unroll
            for (int v = 0; v < 8; ++v) sG[(rbase + v) * G4 + col] = accX[v] + accH[v];
        }
        __syncthreads();

        // ---- elementwise LSTM cell update ----
#pragma unroll
        for (int i = 0; i < 8; ++i) {
            int j = j0 + i;
            float gi = sG[r * G4 + 0 * H + j] + bI[i];
            float gf = sG[r * G4 + 1 * H + j] + bF[i];
            float gg = sG[r * G4 + 2 * H + j] + bG[i];
            float go = sG[r * G4 + 3 * H + j] + bO[i];
            c[i]  = sigm(gf) * c[i] + sigm(gi) * tanhf(gg);
            hv[i] = sigm(go) * tanhf(c[i]);
        }
        bf16_t hb[8];
#pragma unroll
        for (int i = 0; i < 8; ++i) hb[i] = f2bf(hv[i]);
        *(uint4*)(sH + r * H + j0) = *(uint4*)hb;
        *(uint4*)(enc_outs + ((size_t)(b0 + r) * S + t) * H + j0) = *(uint4*)hb;
        // next iteration's top barrier orders sH writes vs. GEMM reads
    }
#pragma unroll
    for (int i = 0; i < 8; ++i) {
        h_final[(b0 + r) * H + j0 + i] = hv[i];
        c_final[(b0 + r) * H + j0 + i] = c[i];
    }
}

// =====================================================================
// K3: ref projections  g_ref/p_ref[b,s,:] = enc_outs[b,s,:] @ W^T + b
// GEMM M=B*S N=128 K=128, both weight mats in LDS, bf16 WMMA.
// =====================================================================
#define SMEM_REF 66560

__global__ __launch_bounds__(256) void refproj_kernel(
    const bf16_t* __restrict__ enc_outs,     // [B*S, H]
    const float* __restrict__ Wg, const float* __restrict__ bg,
    const float* __restrict__ Wp, const float* __restrict__ bp,
    bf16_t* __restrict__ g_ref, bf16_t* __restrict__ p_ref) {
    extern __shared__ char smem[];
    bf16_t* sWg = (bf16_t*)(smem);
    bf16_t* sWp = (bf16_t*)(smem + 32768);
    float*  sbg = (float*) (smem + 65536);
    float*  sbp = (float*) (smem + 66048);

    const int tid = threadIdx.x, wave = tid >> 5, lane = tid & 31;
    for (int i = tid; i < H * H; i += 256) { sWg[i] = f2bf(Wg[i]); sWp[i] = f2bf(Wp[i]); }
    for (int i = tid; i < H; i += 256)     { sbg[i] = bg[i];       sbp[i] = bp[i]; }
    __syncthreads();

    const size_t m0 = (size_t)blockIdx.x * 128 + (size_t)wave * 16;
    v16bf a[4];
#pragma unroll
    for (int ks = 0; ks < 4; ++ks) a[ks] = load_frag(enc_outs + m0 * H, H, ks * 32, lane);

    for (int mat = 0; mat < 2; ++mat) {
        const bf16_t* Wm  = mat ? sWp : sWg;
        const float*  bsv = mat ? sbp : sbg;
        bf16_t*       out = mat ? p_ref : g_ref;
#pragma unroll
        for (int nt = 0; nt < 8; ++nt) {
            v8f acc = {};
#pragma unroll
            for (int ks = 0; ks < 4; ++ks) {
                v16bf bw = load_frag(Wm + nt * 16 * H, H, ks * 32, lane);
                acc = WMMA_BF16(a[ks], bw, acc);
            }
            const int col   = nt * 16 + (lane & 15);
            const int rbase = (lane >> 4) * 8;
            const float bb  = bsv[col];
#pragma unroll
            for (int v = 0; v < 8; ++v)
                out[(m0 + rbase + v) * H + col] = f2bf(acc[v] + bb);
        }
    }
}

// =====================================================================
// K4: decoder state init: mask=1, dec_in = dec_start (bf16)
// =====================================================================
__global__ void init_kernel(const float* __restrict__ dec_start,
                            bf16_t* __restrict__ dec_in,
                            float* __restrict__ mask) {
    int idx = blockIdx.x * blockDim.x + threadIdx.x;
    if (idx < B * H) dec_in[idx] = f2bf(dec_start[idx & (H - 1)]);
    if (idx < B * S) mask[idx] = 1.f;
}

// =====================================================================
// K5: single decoder LSTM cell step (same WMMA structure as encoder)
// =====================================================================
#define SMEM_CELL 303104

__global__ __launch_bounds__(256) void lstm_cell_kernel(
    const bf16_t* __restrict__ x_in,   // [B,H] bf16
    const float* __restrict__ Wih, const float* __restrict__ Whh,
    const float* __restrict__ bih, const float* __restrict__ bhh,
    float* __restrict__ h_buf,         // [B,H] in/out
    float* __restrict__ c_buf,         // [B,H] in/out
    float* __restrict__ query_buf) {   // [B,H] out (query = h)
    extern __shared__ char smem[];
    bf16_t* sWih = (bf16_t*)(smem);
    bf16_t* sWhh = (bf16_t*)(smem + 131072);
    bf16_t* sX   = (bf16_t*)(smem + 262144);
    bf16_t* sH   = (bf16_t*)(smem + 266240);
    float*  sG   = (float*) (smem + 270336);

    const int tid = threadIdx.x, wave = tid >> 5, lane = tid & 31;
    const int b0 = blockIdx.x * 16;
    for (int i = tid; i < G4 * H; i += 256) { sWih[i] = f2bf(Wih[i]); sWhh[i] = f2bf(Whh[i]); }

    const int r  = tid >> 4;
    const int j0 = (tid & 15) * 8;
    *(uint4*)(sX + r * H + j0) = *(const uint4*)(x_in + (size_t)(b0 + r) * H + j0);
    bf16_t hb[8];
#pragma unroll
    for (int i = 0; i < 8; ++i) hb[i] = f2bf(h_buf[(b0 + r) * H + j0 + i]);
    *(uint4*)(sH + r * H + j0) = *(uint4*)hb;
    __syncthreads();

    v16bf ax[4], ah[4];
#pragma unroll
    for (int ks = 0; ks < 4; ++ks) {
        ax[ks] = load_frag(sX, H, ks * 32, lane);
        ah[ks] = load_frag(sH, H, ks * 32, lane);
    }
#pragma unroll
    for (int nt = 0; nt < 4; ++nt) {
        const int n0 = wave * 64 + nt * 16;
        v8f accX = {};
        v8f accH = {};
#pragma unroll
        for (int ks = 0; ks < 4; ++ks) {
            v16bf bx = load_frag(sWih + n0 * H, H, ks * 32, lane);
            v16bf bh = load_frag(sWhh + n0 * H, H, ks * 32, lane);
            accX = WMMA_BF16(ax[ks], bx, accX);
            accH = WMMA_BF16(ah[ks], bh, accH);
        }
        const int col = n0 + (lane & 15), rbase = (lane >> 4) * 8;
#pragma unroll
        for (int v = 0; v < 8; ++v) sG[(rbase + v) * G4 + col] = accX[v] + accH[v];
    }
    __syncthreads();

#pragma unroll
    for (int i = 0; i < 8; ++i) {
        int j = j0 + i;
        float gi = sG[r * G4 + 0 * H + j] + bih[0 * H + j] + bhh[0 * H + j];
        float gf = sG[r * G4 + 1 * H + j] + bih[1 * H + j] + bhh[1 * H + j];
        float gg = sG[r * G4 + 2 * H + j] + bih[2 * H + j] + bhh[2 * H + j];
        float go = sG[r * G4 + 3 * H + j] + bih[3 * H + j] + bhh[3 * H + j];
        float cc = c_buf[(b0 + r) * H + j];
        cc = sigm(gf) * cc + sigm(gi) * tanhf(gg);
        float hh = sigm(go) * tanhf(cc);
        c_buf[(b0 + r) * H + j]     = cc;
        h_buf[(b0 + r) * H + j]     = hh;
        query_buf[(b0 + r) * H + j] = hh;
    }
}

// =====================================================================
// K6: q = (concat(last,query) @ Wc^T + bc) @ Wq^T + bq   (tiny GEMV)
// =====================================================================
__global__ __launch_bounds__(128) void qproj_kernel(
    const float* __restrict__ last, const float* __restrict__ query,
    const float* __restrict__ Wc, const float* __restrict__ bc,
    const float* __restrict__ Wq, const float* __restrict__ bq,
    float* __restrict__ qout) {
    __shared__ float cat[2 * H];
    __shared__ float u[H];
    const int b = blockIdx.x, h = threadIdx.x;
    cat[h]     = last[b * H + h];
    cat[H + h] = query[b * H + h];
    __syncthreads();
    float s = bc[h];
    for (int k = 0; k < 2 * H; ++k) s += cat[k] * Wc[h * 2 * H + k];
    u[h] = s;
    __syncthreads();
    float q = bq[h];
    for (int k = 0; k < H; ++k) q += u[k] * Wq[h * H + k];
    qout[b * H + h] = q;
}

// =====================================================================
// K7: glimpse: logits = V . tanh(q + ref); query' = softmax(logits) @ ref
// one block per batch row; 8 waves stride S; wave32 shfl reductions
// =====================================================================
__global__ __launch_bounds__(256) void glimpse_kernel(
    const float* __restrict__ qv, const bf16_t* __restrict__ ref,
    const float* __restrict__ Vw, float* __restrict__ qout) {
    __shared__ float slog[S];
    __shared__ float sred[8];
    __shared__ float sacc[8][H];
    __shared__ float sq[H];
    __shared__ float smax, ssum;
    const int b = blockIdx.x, tid = threadIdx.x, wave = tid >> 5, lane = tid & 31;
    if (tid < H) sq[tid] = qv[b * H + tid];
    __syncthreads();
    const int h0 = lane * 4;
    float v4[4], q4[4];
#pragma unroll
    for (int j = 0; j < 4; ++j) { v4[j] = Vw[h0 + j]; q4[j] = sq[h0 + j]; }
    const bf16_t* refb = ref + (size_t)b * S * H;
    float wmax = -3.4e38f;
    for (int s = wave; s < S; s += 8) {
        uint2 rr = *(const uint2*)(refb + (size_t)s * H + h0);
        const bf16_t* rp = (const bf16_t*)&rr;
        float t = 0.f;
#pragma unroll
        for (int j = 0; j < 4; ++j) t += tanhf(q4[j] + bf2f(rp[j])) * v4[j];
        for (int off = 16; off; off >>= 1) t += __shfl_xor(t, off, 32);
        if (lane == 0) slog[s] = t;
        wmax = fmaxf(wmax, t);
    }
    if (lane == 0) sred[wave] = wmax;
    __syncthreads();
    if (tid == 0) { float m = sred[0]; for (int w = 1; w < 8; ++w) m = fmaxf(m, sred[w]); smax = m; }
    __syncthreads();
    const float m = smax;
    float acc4[4] = {0, 0, 0, 0}, wsum = 0.f;
    for (int s = wave; s < S; s += 8) {
        float e = __expf(slog[s] - m);
        wsum += e;
        uint2 rr = *(const uint2*)(refb + (size_t)s * H + h0);
        const bf16_t* rp = (const bf16_t*)&rr;
#pragma unroll
        for (int j = 0; j < 4; ++j) acc4[j] += e * bf2f(rp[j]);
    }
    if (lane == 0) sred[wave] = wsum;
#pragma unroll
    for (int j = 0; j < 4; ++j) sacc[wave][h0 + j] = acc4[j];
    __syncthreads();
    if (tid == 0) { float su = 0; for (int w = 0; w < 8; ++w) su += sred[w]; ssum = su; }
    __syncthreads();
    if (tid < H) {
        float a = 0;
        for (int w = 0; w < 8; ++w) a += sacc[w][tid];
        qout[b * H + tid] = a / ssum;
    }
}

// =====================================================================
// K8: pointer logits (no softmax) -> global [B,S]
// =====================================================================
__global__ __launch_bounds__(256) void ptr_logits_kernel(
    const float* __restrict__ qv, const bf16_t* __restrict__ ref,
    const float* __restrict__ Vw, float* __restrict__ logits) {
    __shared__ float sq[H];
    const int b = blockIdx.x, tid = threadIdx.x, wave = tid >> 5, lane = tid & 31;
    if (tid < H) sq[tid] = qv[b * H + tid];
    __syncthreads();
    const int h0 = lane * 4;
    float v4[4], q4[4];
#pragma unroll
    for (int j = 0; j < 4; ++j) { v4[j] = Vw[h0 + j]; q4[j] = sq[h0 + j]; }
    const bf16_t* refb = ref + (size_t)b * S * H;
    for (int s = wave; s < S; s += 8) {
        uint2 rr = *(const uint2*)(refb + (size_t)s * H + h0);
        const bf16_t* rp = (const bf16_t*)&rr;
        float t = 0.f;
#pragma unroll
        for (int j = 0; j < 4; ++j) t += tanhf(q4[j] + bf2f(rp[j])) * v4[j];
        for (int off = 16; off; off >>= 1) t += __shfl_xor(t, off, 32);
        if (lane == 0) logits[b * S + s] = t;
    }
}

// =====================================================================
// K9: mask -> deterministic Gumbel "sampling" -> logp, tours, next dec_in
// =====================================================================
__global__ __launch_bounds__(256) void sample_kernel(
    const float* __restrict__ logits, float* __restrict__ mask,
    const bf16_t* __restrict__ x_emb, bf16_t* __restrict__ dec_in,
    float* __restrict__ out_tours, float* __restrict__ out_logps, int step) {
    __shared__ float sval[256];
    __shared__ int   sidx[256];
    __shared__ float smx[256];
    __shared__ float ssm[256];
    __shared__ int   sptr;
    __shared__ float sM;
    const int b = blockIdx.x, tid = threadIdx.x;
    float bestv = -3.4e38f, mx = -3.4e38f;
    int   besti = 0;
    for (int s = tid; s < S; s += 256) {
        float l  = logits[b * S + s];
        float ml = (mask[b * S + s] > 0.f) ? l : NEGINF;
        mx = fmaxf(mx, ml);
        // deterministic counter-hash -> Gumbel noise (stand-in for jax PRNG)
        unsigned x = (unsigned)(b * 1664525u) ^ (unsigned)(s * 22695477u)
                   ^ (unsigned)(step * 2654435761u) ^ 0x9E3779B9u;
        x ^= x >> 16; x *= 0x7feb352du; x ^= x >> 15; x *= 0x846ca68bu; x ^= x >> 16;
        float u = ((float)(x >> 8) + 0.5f) * (1.f / 16777216.f);
        float g = -__logf(-__logf(u));
        float sc = ml + g;
        if (sc > bestv) { bestv = sc; besti = s; }
    }
    sval[tid] = bestv; sidx[tid] = besti; smx[tid] = mx;
    __syncthreads();
    for (int off = 128; off; off >>= 1) {
        if (tid < off) {
            if (sval[tid + off] > sval[tid]) { sval[tid] = sval[tid + off]; sidx[tid] = sidx[tid + off]; }
            smx[tid] = fmaxf(smx[tid], smx[tid + off]);
        }
        __syncthreads();
    }
    if (tid == 0) { sptr = sidx[0]; sM = smx[0]; }
    __syncthreads();
    const float M = sM;
    float psum = 0.f;
    for (int s = tid; s < S; s += 256) {
        float ml = (mask[b * S + s] > 0.f) ? logits[b * S + s] : NEGINF;
        psum += __expf(ml - M);
    }
    ssm[tid] = psum;
    __syncthreads();
    for (int off = 128; off; off >>= 1) { if (tid < off) ssm[tid] += ssm[tid + off]; __syncthreads(); }
    const int ptr = sptr;
    if (tid == 0) {
        float ml = (mask[b * S + ptr] > 0.f) ? logits[b * S + ptr] : NEGINF;
        out_tours[b * STEPS + step] = (float)ptr;
        out_logps[b * STEPS + step] = ml - M - __logf(ssm[0]);
        mask[b * S + ptr] = 0.f;   // masked for subsequent steps (matches reference)
    }
    __syncthreads();
    if (tid < H) dec_in[b * H + tid] = x_emb[((size_t)b * S + ptr) * H + tid];
}

// =====================================================================
// host orchestration
// =====================================================================
extern "C" void kernel_launch(void* const* d_in, const int* in_sizes, int n_in,
                              void* d_out, int out_size, void* d_ws, size_t ws_size,
                              hipStream_t stream) {
    (void)in_sizes; (void)n_in; (void)out_size; (void)ws_size;
    const float* static_in = (const float*)d_in[0];
    // d_in[1],[4],[5]: dynamic + demb — dead w.r.t. outputs in the reference
    const float* emb_W   = (const float*)d_in[2];
    const float* emb_b   = (const float*)d_in[3];
    const float* enc_Wih = (const float*)d_in[6];
    const float* enc_Whh = (const float*)d_in[7];
    const float* enc_bih = (const float*)d_in[8];
    const float* enc_bhh = (const float*)d_in[9];
    const float* dec_Wih = (const float*)d_in[10];
    const float* dec_Whh = (const float*)d_in[11];
    const float* dec_bih = (const float*)d_in[12];
    const float* dec_bhh = (const float*)d_in[13];
    const float* g_Wc    = (const float*)d_in[14];
    const float* g_bc    = (const float*)d_in[15];
    const float* g_Wq    = (const float*)d_in[16];
    const float* g_bq    = (const float*)d_in[17];
    const float* g_Wref  = (const float*)d_in[18];
    const float* g_bref  = (const float*)d_in[19];
    const float* g_V     = (const float*)d_in[20];
    const float* p_Wc    = (const float*)d_in[21];
    const float* p_bc    = (const float*)d_in[22];
    const float* p_Wq    = (const float*)d_in[23];
    const float* p_bq    = (const float*)d_in[24];
    const float* p_Wref  = (const float*)d_in[25];
    const float* p_bref  = (const float*)d_in[26];
    const float* p_V     = (const float*)d_in[27];
    const float* dec_start = (const float*)d_in[28];

    char*  ws  = (char*)d_ws;
    size_t off = 0;
    auto alloc = [&](size_t bytes) -> void* {
        void* p = ws + off;
        off = (off + bytes + 255) & ~(size_t)255;
        return p;
    };
    bf16_t* x_emb    = (bf16_t*)alloc((size_t)B * S * H * 2);
    bf16_t* enc_outs = (bf16_t*)alloc((size_t)B * S * H * 2);
    bf16_t* g_ref    = (bf16_t*)alloc((size_t)B * S * H * 2);
    bf16_t* p_ref    = (bf16_t*)alloc((size_t)B * S * H * 2);
    float*  h_buf    = (float*) alloc((size_t)B * H * 4);
    float*  c_buf    = (float*) alloc((size_t)B * H * 4);
    float*  query_buf= (float*) alloc((size_t)B * H * 4);
    float*  q_buf    = (float*) alloc((size_t)B * H * 4);
    bf16_t* dec_in   = (bf16_t*)alloc((size_t)B * H * 2);
    float*  mask     = (float*) alloc((size_t)B * S * 4);
    float*  pl       = (float*) alloc((size_t)B * S * 4);

    float* out_tours = (float*)d_out;
    float* out_logps = out_tours + B * STEPS;

    // large dynamic LDS (encoder: 300 KB; cell: 296 KB; refproj: 65 KB)
    (void)hipFuncSetAttribute((const void*)lstm_encoder_kernel,
                              hipFuncAttributeMaxDynamicSharedMemorySize, SMEM_ENC);
    (void)hipFuncSetAttribute((const void*)lstm_cell_kernel,
                              hipFuncAttributeMaxDynamicSharedMemorySize, SMEM_CELL);
    (void)hipFuncSetAttribute((const void*)refproj_kernel,
                              hipFuncAttributeMaxDynamicSharedMemorySize, SMEM_REF);

    embed_kernel<<<(int)(((size_t)B * S * H + 255) / 256), 256, 0, stream>>>(
        static_in, emb_W, emb_b, x_emb);

    lstm_encoder_kernel<<<B / 16, 256, SMEM_ENC, stream>>>(
        x_emb, enc_Wih, enc_Whh, enc_bih, enc_bhh, enc_outs, h_buf, c_buf);

    refproj_kernel<<<(B * S) / 128, 256, SMEM_REF, stream>>>(
        enc_outs, g_Wref, g_bref, p_Wref, p_bref, g_ref, p_ref);

    init_kernel<<<(B * S + 255) / 256, 256, 0, stream>>>(dec_start, dec_in, mask);

    for (int step = 0; step < STEPS; ++step) {
        lstm_cell_kernel<<<B / 16, 256, SMEM_CELL, stream>>>(
            dec_in, dec_Wih, dec_Whh, dec_bih, dec_bhh, h_buf, c_buf, query_buf);
        for (int g = 0; g < GLIMPSES; ++g) {
            qproj_kernel<<<B, H, 0, stream>>>(h_buf, query_buf, g_Wc, g_bc, g_Wq, g_bq, q_buf);
            glimpse_kernel<<<B, 256, 0, stream>>>(q_buf, g_ref, g_V, query_buf);
        }
        qproj_kernel<<<B, H, 0, stream>>>(h_buf, query_buf, p_Wc, p_bc, p_Wq, p_bq, q_buf);
        ptr_logits_kernel<<<B, 256, 0, stream>>>(q_buf, p_ref, p_V, pl);
        sample_kernel<<<B, 256, 0, stream>>>(pl, mask, x_emb, dec_in,
                                             out_tours, out_logps, step);
    }
}
